// MaskTypeProbabilitiesLayer_1279900254761
// MI455X (gfx1250) — matrix operand order repaired and checked by reading, not compiled, not run on hardware
//
#include <hip/hip_runtime.h>

// Problem shape (from reference): inputs (1024, 2048, 11) int32, out (1024,2048,8) -> float 0/1.
#define BATCHN 1024
#define SEQL   2048
#define NFEAT  11

typedef __attribute__((ext_vector_type(16))) _Float16 v16h;
typedef __attribute__((ext_vector_type(8)))  float    v8f;

// Extended-category mask bytes (bit k = output column k):
//  c=0:0x02  c=1:0x06  c=2:0x10  c=3:0xF8  c=4..6:0 (never selected)  c=7:0x80
//  c=8 (no timesign):0x20  c=9 (no tempo):0x40  c=10 (full):0xF8  c>=11:0
#define MASK_LO 0x80000000F8100602ULL  // bytes for categories 0..7
#define MASK_HI 0x0000000000F84020ULL  // bytes for categories 8..15

__global__ __launch_bounds__(256)
void MaskTypeProb_wmma_kernel(const int* __restrict__ in, float* __restrict__ out) {
    __shared__ int s_gt[SEQL];     // staged gt column for this batch row
    __shared__ int s_first5, s_first6;

    const int tid  = threadIdx.x;
    const int b    = blockIdx.x;
    const int lane = tid & 31;
    const int wave = tid >> 5;

    if (tid == 0) { s_first5 = SEQL; s_first6 = SEQL; }

    // ---- Phase 1: stream gt into LDS, track first occurrence of 5 / 6 ----
    const size_t rowbase = (size_t)b * SEQL;
    int m5 = SEQL, m6 = SEQL;
    #pragma unroll
    for (int it = 0; it < 8; ++it) {
        const int l = it * 256 + tid;
        const int v = in[(rowbase + (size_t)l) * NFEAT];  // feature 0 only
        s_gt[l] = v;
        if (v == 5) m5 = min(m5, l);
        if (v == 6) m6 = min(m6, l);
    }
    __syncthreads();
    if (m5 < SEQL) atomicMin(&s_first5, m5);   // ds_min_i32
    if (m6 < SEQL) atomicMin(&s_first6, m6);
    __syncthreads();
    const int first5 = s_first5;
    const int first6 = s_first6;

    // ---- Constant B fragment: B[k][n] = bit n of maskbyte[k]  (32x16, rows>=16 zero) ----
    // B, contiguous k order used here: lane n (0..15) holds halves[k] = B[k][n]; lanes 16..31 hold K=16..31 (all zero).
    // Only columns n<8 are nonzero -> only lanes 0..7 carry data.
    v16h bfrag;
    #pragma unroll
    for (int j = 0; j < 16; ++j) {
        const unsigned mb = (unsigned)(((j < 8) ? (MASK_LO >> (8 * j))
                                                : (MASK_HI >> (8 * (j - 8)))) & 0xFFu);
        bfrag[j] = (_Float16)(float)((lane < 8) ? ((mb >> lane) & 1u) : 0u);
    }

    // A, contiguous k order used here (16x32): lanes 0..15 = row M=lane, halves 0..15 -> K=0..15;
    //                          lanes 16..31 = row M=lane-16, halves 0..15 -> K=16..31.
    // lanes 16..31 map to K>=16, always zero here (categories < 16).
    const int kbase = (lane & 16) ? 8 : 0;
    const int mrow  = lane & 15;
    const int col   = lane & 15;            // D: N = lane & 15 (valid output col if < 8)
    const int hi8   = (lane >> 4) << 3;     // D: M = r + 8*(lane>>4)

    // ---- Phase 2: 16 WMMA tiles per wave, 16 positions each ----
    #pragma unroll 1
    for (int t = 0; t < 16; ++t) {
        const int l0  = wave * 256 + t * 16;
        const int pos = l0 + mrow;
        const int v   = s_gt[pos];
        const int p   = min(pos + 1, SEQL - 1);

        int cat;
        if (v >= 4 && v <= 6)
            cat = (p < first5) ? 8 : ((p < first6) ? 9 : 10);
        else
            cat = v;  // v in {0,1,2,3,7}

        v16h a;
        #pragma unroll
        for (int j = 0; j < 8; ++j) {
            a[j]     = (kbase == 0 && cat == j) ? (_Float16)1.0f : (_Float16)0.0f;
            a[j + 8] = (kbase == 0 && cat == j + 8) ? (_Float16)1.0f : (_Float16)0.0f;
        }

        v8f c = {};
        // D[m][n] = onehot(cat[m]) . TABLE[:,n]  -> exact 0.0/1.0 mask floats
        v8f d = __builtin_amdgcn_wmma_f32_16x16x32_f16(
            /*neg_a=*/false, a, /*neg_b=*/false, bfrag,
            /*c_mod=*/(short)0, c, /*reuse_a=*/false, /*reuse_b=*/false);

        // Store: VGPR r holds (M = r + hi8, N = col); only N < 8 exists in the output.
        if (col < 8) {
            const size_t obase = (rowbase + (size_t)(l0 + hi8)) * 8 + (size_t)col;
            #pragma unroll
            for (int r = 0; r < 8; ++r)
                __builtin_nontemporal_store(d[r], &out[obase + (size_t)r * 8]);
        }
    }
}

extern "C" void kernel_launch(void* const* d_in, const int* in_sizes, int n_in,
                              void* d_out, int out_size, void* d_ws, size_t ws_size,
                              hipStream_t stream) {
    (void)in_sizes; (void)n_in; (void)out_size; (void)d_ws; (void)ws_size;
    const int* in = (const int*)d_in[0];
    float* out = (float*)d_out;
    MaskTypeProb_wmma_kernel<<<dim3(BATCHN), dim3(256), 0, stream>>>(in, out);
}
